// FineDeformableTransformer_73873437491493
// MI455X (gfx1250) — matrix operand order repaired and checked by reading, compiled, not run on hardware
//
#include <hip/hip_runtime.h>

// ---------------------------------------------------------------------------
// FineDeformableTransformer forward for MI455X (gfx1250, wave32, WMMA).
// GEMMs run on v_wmma_f32_16x16x32_f16 (f16 in, f32 accumulate); weights are
// pre-packed into B-fragment lane order so the inner loop is 3 vector loads +
// 1 WMMA. Deformable sampling / softmax / layernorm are scalar f32 kernels.
// ---------------------------------------------------------------------------

typedef __attribute__((ext_vector_type(16))) _Float16 v16h;
typedef __attribute__((ext_vector_type(8)))  _Float16 v8h;
typedef __attribute__((ext_vector_type(8)))  float    v8f;

#define DM      256          // d_model
#define DFFN    1024
#define NHEADS  8
#define HDIM    32
#define LTOT    7902         // sum of level sizes
#define LQ      7934         // LTOT + 32 vt tokens
#define LQPAD   7936         // padded to multiple of 16 for WMMA M tiles
#define BB      2            // batch

// level tables: SHAPES = [(12,24,24),(6,12,12),(3,6,6),(2,3,3)]
#define LVL_TABLES \
    const int ST[4]={0,6912,7776,7884}; \
    const int DD[4]={12,6,3,2}; \
    const int HH[4]={24,12,6,3}; \
    const int WW[4]={24,12,6,3};

struct Corners { int c[32]; };

// ---------------------------------------------------------------------------
// Token assembly: x = concat(src_f, sel_vt); pos_cat = concat(pos_f+lvl_emb, 0)
// ---------------------------------------------------------------------------
__global__ void build_tokens_kernel(
    const float* __restrict__ s0, const float* __restrict__ s1,
    const float* __restrict__ s2, const float* __restrict__ s3,
    const float* __restrict__ p0, const float* __restrict__ p1,
    const float* __restrict__ p2, const float* __restrict__ p3,
    const float* __restrict__ lvl_emb, const float* __restrict__ all_vt,
    const int* __restrict__ vt_pos,
    float* __restrict__ x, float* __restrict__ pos_cat)
{
    size_t idx = (size_t)blockIdx.x * blockDim.x + threadIdx.x;
    const size_t total = (size_t)BB * LQPAD * DM;
    if (idx >= total) return;
    int c = (int)(idx % DM);
    int l = (int)((idx / DM) % LQPAD);
    int b = (int)(idx / ((size_t)DM * LQPAD));
    float xv = 0.f, pv = 0.f;
    if (l < LTOT) {
        LVL_TABLES
        const int NN[4] = {6912, 864, 108, 18};
        int lvl = (l < 6912) ? 0 : (l < 7776) ? 1 : (l < 7884) ? 2 : 3;
        int sp = l - ST[lvl];
        int n  = NN[lvl];
        const float* S = (lvl==0)?s0:(lvl==1)?s1:(lvl==2)?s2:s3;
        const float* P = (lvl==0)?p0:(lvl==1)?p1:(lvl==2)?p2:p3;
        size_t o = ((size_t)b * DM + c) * n + sp;       // (B,256,D,H,W) layout
        xv = S[o];
        pv = P[o] + lvl_emb[lvl * DM + c];
        (void)DD; (void)HH; (void)WW;
    } else if (l < LQ) {
        int j = l - LTOT;                 // vt token: j = lvl*8 + i
        int lvl = j >> 3, i = j & 7;
        int vp = vt_pos[b * 8 + i];
        xv = all_vt[((size_t)lvl * 75 + vp) * DM + c];
        pv = 0.f;
    }
    x[idx] = xv;
    pos_cat[idx] = pv;
}

// ---------------------------------------------------------------------------
// Reference points. Masks are all-false => valid_ratios == 1, so the point is
// batch- and level-independent: [ (d+.5)/D, (x+.5)/W, (y+.5)/H ].
// vt tokens reuse the reference's corner indices verbatim (they index the
// concatenated ref with per-level *local* offsets, matching the reference).
// ---------------------------------------------------------------------------
__device__ inline void ref_for_token(int l, float r[3])
{
    LVL_TABLES
    int lvl = (l < 6912) ? 0 : (l < 7776) ? 1 : (l < 7884) ? 2 : 3;
    int sp = l - ST[lvl];
    int D = DD[lvl], H = HH[lvl], W = WW[lvl];
    int d = sp / (H * W);
    int rem = sp % (H * W);
    int y = rem / W, xx = rem % W;
    r[0] = (d + 0.5f) / D;
    r[1] = (xx + 0.5f) / W;
    r[2] = (y + 0.5f) / H;
}

__global__ void build_ref_kernel(float* __restrict__ ref, Corners cs)
{
    int l = blockIdx.x * blockDim.x + threadIdx.x;
    if (l >= LQPAD) return;
    float r[3] = {0.5f, 0.5f, 0.5f};
    if (l < LTOT)      ref_for_token(l, r);
    else if (l < LQ)   ref_for_token(cs.c[l - LTOT], r);
    ref[l*3+0] = r[0]; ref[l*3+1] = r[1]; ref[l*3+2] = r[2];
}

// ---------------------------------------------------------------------------
// Per-layer q/v prep: q16 = f16(x + pos_cat), v16 = f16(x)
// ---------------------------------------------------------------------------
__global__ void prep_qv_kernel(const float* __restrict__ x,
                               const float* __restrict__ pos,
                               _Float16* __restrict__ q16,
                               _Float16* __restrict__ v16)
{
    size_t i = (size_t)blockIdx.x * blockDim.x + threadIdx.x;
    const size_t total = (size_t)BB * LQPAD * DM;
    if (i >= total) return;
    float xv = x[i];
    q16[i] = (_Float16)(xv + pos[i]);
    v16[i] = (_Float16)xv;
}

// ---------------------------------------------------------------------------
// Pack a (K,N) row-major f32 weight into WMMA B-fragment order:
//   Wpk[(nt*(K/32)+kt)*512 + lane*16 + e] = W[kt*32 + 16*(e>>3) + 8*(lane>>4) + (e&7)][nt*16 + (lane&15)]
// so a GEMM lane fetches its whole 16-half B fragment with one contiguous load.
// ---------------------------------------------------------------------------
__global__ void pack_w_kernel(const float* __restrict__ W,
                              _Float16* __restrict__ Wpk, int K, int N)
{
    int idx = blockIdx.x * blockDim.x + threadIdx.x;
    int total = (N >> 4) * (K >> 5) * 512;
    if (idx >= total) return;
    int e    = idx & 15;
    int lane = (idx >> 4) & 31;
    int t    = idx >> 9;
    int ktc  = K >> 5;
    int nt = t / ktc, kt = t % ktc;
    int n = nt * 16 + (lane & 15);
    int k = kt * 32 + ((e >> 3) * 16) + ((lane >> 4) * 8) + (e & 7);
    Wpk[idx] = (_Float16)W[(size_t)k * N + n];
}

// ---------------------------------------------------------------------------
// WMMA GEMM: C[M,N] = A[M,K] (f16, row-major) x Wpk (packed f16) + bias.
// One 16x16 output tile per wave32; K consumed 32/step.
// EPI: 0 -> f32 store, 1 -> relu + f16 store.
// ---------------------------------------------------------------------------
template <int EPI>
__global__ void gemm_wmma_kernel(const _Float16* __restrict__ A,
                                 const _Float16* __restrict__ Wpk,
                                 const float* __restrict__ bias,
                                 void* __restrict__ Out,
                                 int M, int N, int K)
{
    const int lane = threadIdx.x & 31;
    const int wave = threadIdx.x >> 5;
    const int mt = blockIdx.x;
    const int nt = blockIdx.y * 4 + wave;
    if (nt * 16 >= N) return;                 // wave-uniform: EXEC stays all-1s

    const int  half = lane >> 4;
    const int  m    = mt * 16 + (lane & 15);
    const int  ktc  = K >> 5;
    const _Float16* __restrict__ arow  = A + (size_t)m * K;
    const _Float16* __restrict__ wbase = Wpk + ((size_t)nt * ktc) * 512 + lane * 16;

    v8f acc = {};
    for (int kt = 0; kt < ktc; ++kt) {
        const int kb = kt * 32;
        // A fragment (16-bit A 16x32 layout): lane's K = {8h..8h+7} U {16+8h..16+8h+7}
        v8h lo = *(const v8h*)(arow + kb + 8 * half);
        v8h hi = *(const v8h*)(arow + kb + 16 + 8 * half);
        v16h a;
#pragma unroll
        for (int i = 0; i < 8; ++i) { a[i] = lo[i]; a[i + 8] = hi[i]; }
        // B fragment: one contiguous 32-byte strip per lane (pre-packed)
        v16h bfrag = *(const v16h*)(wbase + kt * 512);
        if (kt + 1 < ktc) __builtin_prefetch(wbase + (kt + 1) * 512, 0, 1);
        acc = __builtin_amdgcn_wmma_f32_16x16x32_f16(
            false, a, false, bfrag, (short)0, acc, false, false);
    }

    const int col = nt * 16 + (lane & 15);
    const float bcol = bias[col];
#pragma unroll
    for (int r = 0; r < 8; ++r) {
        int row = mt * 16 + r + 8 * half;     // C layout: VGPR r -> M = r + 8*half
        float v = acc[r] + bcol;
        if (EPI == 0) {
            ((float*)Out)[(size_t)row * N + col] = v;
        } else {
            v = fmaxf(v, 0.f);
            ((_Float16*)Out)[(size_t)row * N + col] = (_Float16)v;
        }
    }
}

// ---------------------------------------------------------------------------
// Softmax over the 16 (level,point) logits per (token, head), in place.
// ---------------------------------------------------------------------------
__global__ void softmax16_kernel(float* __restrict__ attn, int nrows)
{
    int r = blockIdx.x * blockDim.x + threadIdx.x;
    if (r >= nrows) return;
    float* p = attn + (size_t)r * 16;
    float mx = p[0];
#pragma unroll
    for (int i = 1; i < 16; ++i) mx = fmaxf(mx, p[i]);
    float s = 0.f;
#pragma unroll
    for (int i = 0; i < 16; ++i) { float e = __expf(p[i] - mx); p[i] = e; s += e; }
    float inv = 1.f / s;
#pragma unroll
    for (int i = 0; i < 16; ++i) p[i] *= inv;
}

// ---------------------------------------------------------------------------
// Deformable trilinear sampling + attention reduce.
// Block = one (b,l) token, 256 threads: wave = head, lane = channel.
// ---------------------------------------------------------------------------
__global__ void sample_kernel(const float* __restrict__ value,
                              const float* __restrict__ offb,
                              const float* __restrict__ attn,
                              const float* __restrict__ ref,
                              _Float16* __restrict__ samp16)
{
    int bl = blockIdx.x;                 // b*LQPAD + l
    int l = bl % LQPAD;
    int b = bl / LQPAD;
    int h = threadIdx.x >> 5;
    int c = threadIdx.x & 31;
    size_t row = (size_t)bl;
    if (l >= LQ) { samp16[row * DM + h * HDIM + c] = (_Float16)0.f; return; }

    LVL_TABLES
    float r0 = ref[l*3+0], r1 = ref[l*3+1], r2 = ref[l*3+2];
    const float* orow = offb + row * 384 + h * 48;
    const float* arow = attn + row * 128 + h * 16;
    const float* vb   = value + (size_t)b * LQPAD * DM;

    float acc = 0.f;
#pragma unroll
    for (int lvl = 0; lvl < 4; ++lvl) {
        int D = DD[lvl], H = HH[lvl], W = WW[lvl], st = ST[lvl];
#pragma unroll
        for (int p = 0; p < 4; ++p) {
            float o0 = orow[lvl*12 + p*3 + 0];
            float o1 = orow[lvl*12 + p*3 + 1];
            float o2 = orow[lvl*12 + p*3 + 2];
            float pd = (r0 + o0 / D) * D - 0.5f;
            float px = (r1 + o1 / W) * W - 0.5f;
            float py = (r2 + o2 / H) * H - 0.5f;
            float d0f = floorf(pd), x0f = floorf(px), y0f = floorf(py);
            float fd = pd - d0f, fx = px - x0f, fy = py - y0f;
            int d0 = (int)d0f, x0 = (int)x0f, y0 = (int)y0f;
            float s = 0.f;
#pragma unroll
            for (int dd = 0; dd < 2; ++dd)
#pragma unroll
            for (int dy = 0; dy < 2; ++dy)
#pragma unroll
            for (int dx = 0; dx < 2; ++dx) {
                int di = d0 + dd, yi = y0 + dy, xi = x0 + dx;
                if (di >= 0 && di < D && yi >= 0 && yi < H && xi >= 0 && xi < W) {
                    float w = (dd ? fd : 1.f - fd) * (dy ? fy : 1.f - fy) *
                              (dx ? fx : 1.f - fx);
                    int idx = (di * H + yi) * W + xi;
                    s += w * vb[(size_t)(st + idx) * DM + h * HDIM + c];
                }
            }
            acc += arow[lvl*4 + p] * s;
        }
    }
    samp16[row * DM + h * HDIM + c] = (_Float16)acc;
}

// ---------------------------------------------------------------------------
// LayerNorm over 256 channels: y = LN(A + R)*g + b -> Y32 (and optional f16).
// One token per 256-thread block; LDS tree reductions.
// ---------------------------------------------------------------------------
__global__ void layernorm_kernel(const float* __restrict__ A,
                                 const float* __restrict__ R,
                                 const float* __restrict__ g,
                                 const float* __restrict__ beta,
                                 float* __restrict__ Y32,
                                 _Float16* __restrict__ Y16)
{
    __shared__ float sh[DM];
    int t = threadIdx.x;
    size_t row = (size_t)blockIdx.x;
    float v = A[row * DM + t] + R[row * DM + t];
    sh[t] = v;
    __syncthreads();
    for (int s = DM / 2; s > 0; s >>= 1) {
        if (t < s) sh[t] += sh[t + s];
        __syncthreads();
    }
    float mu = sh[0] * (1.f / DM);
    __syncthreads();
    float d = v - mu;
    sh[t] = d * d;
    __syncthreads();
    for (int s = DM / 2; s > 0; s >>= 1) {
        if (t < s) sh[t] += sh[t + s];
        __syncthreads();
    }
    float var = sh[0] * (1.f / DM);
    float y = d * rsqrtf(var + 1e-5f) * g[t] + beta[t];
    Y32[row * DM + t] = y;
    if (Y16) Y16[row * DM + t] = (_Float16)y;
}

// ---------------------------------------------------------------------------
// Emit output: first LTOT tokens of the state.
// ---------------------------------------------------------------------------
__global__ void emit_kernel(const float* __restrict__ x, float* __restrict__ out)
{
    size_t idx = (size_t)blockIdx.x * blockDim.x + threadIdx.x;
    const size_t total = (size_t)BB * LTOT * DM;
    if (idx >= total) return;
    int c = (int)(idx % DM);
    int l = (int)((idx / DM) % LTOT);
    int b = (int)(idx / ((size_t)DM * LTOT));
    out[idx] = x[((size_t)b * LQPAD + l) * DM + c];
}

// ---------------------------------------------------------------------------
// Host orchestration
// ---------------------------------------------------------------------------
static inline size_t align256(size_t v) { return (v + 255) & ~(size_t)255; }

extern "C" void kernel_launch(void* const* d_in, const int* in_sizes, int n_in,
                              void* d_out, int out_size, void* d_ws, size_t ws_size,
                              hipStream_t stream)
{
    (void)in_sizes; (void)n_in; (void)out_size;
    // Input order: src0,pos0,mask0, src1,pos1,mask1, ... vt_pos, then params.
    const float* src[4] = {(const float*)d_in[0], (const float*)d_in[3],
                           (const float*)d_in[6], (const float*)d_in[9]};
    const float* pos[4] = {(const float*)d_in[1], (const float*)d_in[4],
                           (const float*)d_in[7], (const float*)d_in[10]};
    const int*   vt_pos   = (const int*)d_in[12];
    const float* lvl_emb  = (const float*)d_in[13];
    const float* all_vt   = (const float*)d_in[14];
    const float* Wv    = (const float*)d_in[15]; const float* bv    = (const float*)d_in[16];
    const float* Woff  = (const float*)d_in[17]; const float* boff  = (const float*)d_in[18];
    const float* Wattn = (const float*)d_in[19]; const float* battn = (const float*)d_in[20];
    const float* Wout  = (const float*)d_in[21]; const float* bout  = (const float*)d_in[22];
    const float* ln1g  = (const float*)d_in[23]; const float* ln1b  = (const float*)d_in[24];
    const float* W1    = (const float*)d_in[25]; const float* b1    = (const float*)d_in[26];
    const float* W2    = (const float*)d_in[27]; const float* b2    = (const float*)d_in[28];
    const float* ln2g  = (const float*)d_in[29]; const float* ln2b  = (const float*)d_in[30];

    // Workspace carve-up (~180 MB total).
    const size_t TOK  = (size_t)BB * LQPAD;          // padded token rows
    char* base = (char*)d_ws;
    size_t off = 0;
    auto take = [&](size_t bytes) { char* p = base + off; off += align256(bytes); return p; };
    float*    refb    = (float*)   take((size_t)LQPAD * 3 * 4);
    float*    pos_cat = (float*)   take(TOK * DM * 4);
    float*    x       = (float*)   take(TOK * DM * 4);
    float*    x2      = (float*)   take(TOK * DM * 4);
    float*    value   = (float*)   take(TOK * DM * 4);
    float*    tmp     = (float*)   take(TOK * DM * 4);
    float*    offbuf  = (float*)   take(TOK * 384 * 4);
    float*    attnbuf = (float*)   take(TOK * 128 * 4);
    _Float16* q16     = (_Float16*)take(TOK * DM * 2);
    _Float16* v16     = (_Float16*)take(TOK * DM * 2);
    _Float16* s16     = (_Float16*)take(TOK * DM * 2);
    _Float16* samp16  = (_Float16*)take(TOK * DM * 2);
    _Float16* h16     = (_Float16*)take(TOK * DFFN * 2);
    _Float16* wpk     = (_Float16*)take((size_t)(DFFN / 32) * (DM / 16) * 512 * 2);
    if (off > ws_size) return;   // insufficient scratch (deterministic no-op)

    // Corner indices exactly as the reference builds them (per-level local
    // offsets used as global ref indices, with W_=h, H_=w swapped).
    const int SH[4][3] = {{12,24,24},{6,12,12},{3,6,6},{2,3,3}};
    Corners cs;
    int ci = 0;
    for (int s = 0; s < 4; ++s) {
        int D_ = SH[s][0], W_ = SH[s][1], H_ = SH[s][2];
        int v0 = H_ * W_ * (D_ - 1);
        int loc[8] = {0, H_ - 1, H_ * (W_ - 1) - 1, H_ * W_ - 1,
                      v0, v0 + H_ - 1, v0 + H_ * (W_ - 1) - 1, v0 + H_ * W_ - 1};
        for (int j = 0; j < 8; ++j) cs.c[ci++] = loc[j];
    }

    const int M = (int)TOK;                       // 15872 rows (mult of 16)
    const size_t nelem = TOK * DM;
    dim3 thr(256);
    dim3 grdTok((unsigned)((nelem + 255) / 256));

    build_tokens_kernel<<<grdTok, thr, 0, stream>>>(
        src[0], src[1], src[2], src[3], pos[0], pos[1], pos[2], pos[3],
        lvl_emb, all_vt, vt_pos, x, pos_cat);
    build_ref_kernel<<<(LQPAD + 255) / 256, thr, 0, stream>>>(refb, cs);

    auto pack = [&](const float* W, int K, int N) {
        int total = (N >> 4) * (K >> 5) * 512;
        pack_w_kernel<<<(total + 255) / 256, thr, 0, stream>>>(W, wpk, K, N);
    };
    auto gemm_f32 = [&](const _Float16* A, const float* bias, float* out, int N, int K) {
        dim3 g(M / 16, (N + 63) / 64);
        gemm_wmma_kernel<0><<<g, dim3(128), 0, stream>>>(A, wpk, bias, out, M, N, K);
    };
    auto gemm_relu16 = [&](const _Float16* A, const float* bias, _Float16* out, int N, int K) {
        dim3 g(M / 16, (N + 63) / 64);
        gemm_wmma_kernel<1><<<g, dim3(128), 0, stream>>>(A, wpk, bias, out, M, N, K);
    };

    for (int li = 0; li < 2; ++li) {
        const float* Wv_l    = Wv    + (size_t)li * DM * DM;
        const float* bv_l    = bv    + (size_t)li * DM;
        const float* Woff_l  = Woff  + (size_t)li * DM * 384;
        const float* boff_l  = boff  + (size_t)li * 384;
        const float* Wattn_l = Wattn + (size_t)li * DM * 128;
        const float* battn_l = battn + (size_t)li * 128;
        const float* Wout_l  = Wout  + (size_t)li * DM * DM;
        const float* bout_l  = bout  + (size_t)li * DM;
        const float* W1_l    = W1    + (size_t)li * DM * DFFN;
        const float* b1_l    = b1    + (size_t)li * DFFN;
        const float* W2_l    = W2    + (size_t)li * DFFN * DM;
        const float* b2_l    = b2    + (size_t)li * DM;

        prep_qv_kernel<<<grdTok, thr, 0, stream>>>(x, pos_cat, q16, v16);

        pack(Wv_l, DM, DM);      gemm_f32(v16, bv_l, value, DM, DM);
        pack(Woff_l, DM, 384);   gemm_f32(q16, boff_l, offbuf, 384, DM);
        pack(Wattn_l, DM, 128);  gemm_f32(q16, battn_l, attnbuf, 128, DM);

        int srows = (int)(TOK * NHEADS);
        softmax16_kernel<<<(srows + 255) / 256, thr, 0, stream>>>(attnbuf, srows);

        sample_kernel<<<(unsigned)TOK, thr, 0, stream>>>(value, offbuf, attnbuf,
                                                         refb, samp16);

        pack(Wout_l, DM, DM);    gemm_f32(samp16, bout_l, tmp, DM, DM);

        // s = LN(v + attn_out); keep f32 state in x2, f16 copy for FFN
        layernorm_kernel<<<(unsigned)TOK, thr, 0, stream>>>(
            tmp, x, ln1g + (size_t)li * DM, ln1b + (size_t)li * DM, x2, s16);

        pack(W1_l, DM, DFFN);    gemm_relu16(s16, b1_l, h16, DFFN, DM);
        pack(W2_l, DFFN, DM);    gemm_f32(h16, b2_l, tmp, DM, DFFN);

        // state = LN(s + ffn)
        layernorm_kernel<<<(unsigned)TOK, thr, 0, stream>>>(
            tmp, x2, ln2g + (size_t)li * DM, ln2b + (size_t)li * DM, x, nullptr);
    }

    const size_t outN = (size_t)BB * LTOT * DM;
    emit_kernel<<<(unsigned)((outN + 255) / 256), thr, 0, stream>>>(x, (float*)d_out);
}
